// CoLightMultiHeadGAT_80564996538676
// MI455X (gfx1250) — matrix operand order repaired and checked by reading, not compiled, not run on hardware
//
#include <hip/hip_runtime.h>
#include <hip/hip_bf16.h>
#include <math.h>

// CDNA5 wave32 WMMA types
typedef __attribute__((ext_vector_type(16))) _Float16 v16h;
typedef __attribute__((ext_vector_type(8)))  _Float16 v8h;
typedef __attribute__((ext_vector_type(8)))  float    v8f;

#define DM 1024
#define BM 128
#define BN 64
#define BK 32
#define MEGA (1024 * 1024)

// ---------------------------------------------------------------------------
// Prep 1: f32 -> f16 row-major copy of embedded ([1024,1024]).
// ---------------------------------------------------------------------------
__global__ __launch_bounds__(256) void cvt_f32_to_f16(
    const float* __restrict__ src, _Float16* __restrict__ dst)
{
    const int e = (blockIdx.x * 256 + threadIdx.x) * 8;
    const float4 f0 = *(const float4*)(src + e);
    const float4 f1 = *(const float4*)(src + e + 4);
    v8h o;
    o[0] = (_Float16)f0.x; o[1] = (_Float16)f0.y;
    o[2] = (_Float16)f0.z; o[3] = (_Float16)f0.w;
    o[4] = (_Float16)f1.x; o[5] = (_Float16)f1.y;
    o[6] = (_Float16)f1.z; o[7] = (_Float16)f1.w;
    *(v8h*)(dst + e) = o;
}

// ---------------------------------------------------------------------------
// Prep 2: W [k][n] f32 -> Wt [n][k] f16 (32x32 LDS tile transpose).
// ---------------------------------------------------------------------------
__global__ __launch_bounds__(256) void transpose_to_f16(
    const float* __restrict__ W0, const float* __restrict__ W1,
    const float* __restrict__ W2,
    _Float16* __restrict__ T0, _Float16* __restrict__ T1,
    _Float16* __restrict__ T2)
{
    const float* W; _Float16* T;
    switch (blockIdx.z) {
        case 0:  W = W0; T = T0; break;
        case 1:  W = W1; T = T1; break;
        default: W = W2; T = T2; break;
    }
    __shared__ float tile[32][33];
    const int k0 = blockIdx.y * 32, n0 = blockIdx.x * 32;
    const int tx = threadIdx.x & 31, ty = threadIdx.x >> 5;
    #pragma unroll
    for (int r = 0; r < 32; r += 8)
        tile[ty + r][tx] = W[(size_t)(k0 + ty + r) * DM + n0 + tx];
    __syncthreads();
    #pragma unroll
    for (int r = 0; r < 32; r += 8)
        T[(size_t)(n0 + ty + r) * DM + k0 + tx] = (_Float16)tile[tx][ty + r];
}

// ---------------------------------------------------------------------------
// GEMM: Y = relu(X @ W + b), f16 in, f32 WMMA accum, f16 out.
// Block: 128 thr = 4 waves; tile 128x64; wave w owns rows [w*32, w*32+32).
// Double-buffered fragment-order LDS: ONE barrier per K-chunk; next chunk's
// staging (global b128 -> ds_store_b128) overlaps current chunk's WMMAs.
//   A elem (m,k): lane=(m&15)|(((k>>3)&1)<<4), slot=(k&7)|((k>>4)<<3)
//   B elem (n,k): lane=(n&15)|(((k>>3)&1)<<4), slot=(k&7)|((k>>4)<<3)
// ---------------------------------------------------------------------------
__global__ __launch_bounds__(128) void gemm_bias_relu_wmma(
    const _Float16* __restrict__ Xh,
    const _Float16* __restrict__ T0, const float* __restrict__ B0,
    const _Float16* __restrict__ T1, const float* __restrict__ B1,
    const _Float16* __restrict__ T2, const float* __restrict__ B2,
    _Float16* __restrict__ Y0, _Float16* __restrict__ Y1,
    _Float16* __restrict__ Y2)
{
    const _Float16* Wt; const float* bias; _Float16* Y;
    switch (blockIdx.z) {
        case 0:  Wt = T0; bias = B0; Y = Y0; break;
        case 1:  Wt = T1; bias = B1; Y = Y1; break;
        default: Wt = T2; bias = B2; Y = Y2; break;
    }

    // [buffer][tile][lane][slot] : A 2x8KB, B 2x4KB = 24KB LDS
    __shared__ __attribute__((aligned(32))) _Float16 aF[2][8][32][16];
    __shared__ __attribute__((aligned(32))) _Float16 bF[2][4][32][16];

    const int tid  = threadIdx.x;
    const int wave = tid >> 5;
    const int lane = tid & 31;
    const int m0   = blockIdx.y * BM;
    const int n0   = blockIdx.x * BN;

    // staging coordinates (fixed per thread)
    const int ea_r[4] = { (tid + 0) >> 2, (tid + 128) >> 2,
                          (tid + 256) >> 2, (tid + 384) >> 2 };
    const int ea_k = (tid & 3) * 8;

    v8f acc[2][4] = {};

    // ---- stage chunk 0 into buffer 0 ----
    #pragma unroll
    for (int i = 0; i < 4; ++i) {
        const int r = ea_r[i], k = ea_k;
        v8h v = *(const v8h*)(Xh + (size_t)(m0 + r) * DM + k);
        *(v8h*)&aF[0][r >> 4][(r & 15) | (((k >> 3) & 1) << 4)][(k >> 4) << 3] = v;
    }
    #pragma unroll
    for (int i = 0; i < 2; ++i) {
        const int e = tid + i * 128;
        const int n = e >> 2, k = (e & 3) * 8;
        v8h v = *(const v8h*)(Wt + (size_t)(n0 + n) * DM + k);
        *(v8h*)&bF[0][n >> 4][(n & 15) | (((k >> 3) & 1) << 4)][(k >> 4) << 3] = v;
    }
    __syncthreads();

    for (int c = 0; c < DM / BK; ++c) {
        const int cur = c & 1, nxt = cur ^ 1;

        // ---- prefetch chunk c+1 into the other buffer (no wait needed) ----
        if (c + 1 < DM / BK) {
            const int kc = (c + 1) * BK;
            #pragma unroll
            for (int i = 0; i < 4; ++i) {
                const int r = ea_r[i], k = ea_k;
                v8h v = *(const v8h*)(Xh + (size_t)(m0 + r) * DM + kc + k);
                *(v8h*)&aF[nxt][r >> 4][(r & 15) | (((k >> 3) & 1) << 4)]
                       [(k >> 4) << 3] = v;
            }
            #pragma unroll
            for (int i = 0; i < 2; ++i) {
                const int e = tid + i * 128;
                const int n = e >> 2, k = (e & 3) * 8;
                v8h v = *(const v8h*)(Wt + (size_t)(n0 + n) * DM + kc + k);
                *(v8h*)&bF[nxt][n >> 4][(n & 15) | (((k >> 3) & 1) << 4)][(k >> 4) << 3] = v;
            }
        }

        // ---- load ALL fragments first (in-order LDS returns let the
        //      compiler use partial dscnt waits), then the WMMA burst ----
        const v16h a0  = *(const v16h*)&aF[cur][wave * 2 + 0][lane][0];
        const v16h a1  = *(const v16h*)&aF[cur][wave * 2 + 1][lane][0];
        const v16h bb0 = *(const v16h*)&bF[cur][0][lane][0];
        const v16h bb1 = *(const v16h*)&bF[cur][1][lane][0];
        const v16h bb2 = *(const v16h*)&bF[cur][2][lane][0];
        const v16h bb3 = *(const v16h*)&bF[cur][3][lane][0];

        acc[0][0] = __builtin_amdgcn_wmma_f32_16x16x32_f16(
            false, a0, false, bb0, (short)0, acc[0][0], false, false);
        acc[1][0] = __builtin_amdgcn_wmma_f32_16x16x32_f16(
            false, a1, false, bb0, (short)0, acc[1][0], false, false);
        acc[0][1] = __builtin_amdgcn_wmma_f32_16x16x32_f16(
            false, a0, false, bb1, (short)0, acc[0][1], false, false);
        acc[1][1] = __builtin_amdgcn_wmma_f32_16x16x32_f16(
            false, a1, false, bb1, (short)0, acc[1][1], false, false);
        acc[0][2] = __builtin_amdgcn_wmma_f32_16x16x32_f16(
            false, a0, false, bb2, (short)0, acc[0][2], false, false);
        acc[1][2] = __builtin_amdgcn_wmma_f32_16x16x32_f16(
            false, a1, false, bb2, (short)0, acc[1][2], false, false);
        acc[0][3] = __builtin_amdgcn_wmma_f32_16x16x32_f16(
            false, a0, false, bb3, (short)0, acc[0][3], false, false);
        acc[1][3] = __builtin_amdgcn_wmma_f32_16x16x32_f16(
            false, a1, false, bb3, (short)0, acc[1][3], false, false);

        // one barrier per chunk: readers of `cur` done + writers of `nxt` done
        __syncthreads();
    }

    // ---- epilogue: D layout lane n=lane&15, VGPR r: m = r + 8*(lane>>4)
    const int nl = lane & 15, hi = lane >> 4;
    #pragma unroll
    for (int t = 0; t < 4; ++t) {
        const int gcol = n0 + t * 16 + nl;
        const float bv = bias[gcol];
        #pragma unroll
        for (int rt = 0; rt < 2; ++rt) {
            #pragma unroll
            for (int r = 0; r < 8; ++r) {
                const int grow = m0 + (wave * 2 + rt) * 16 + r + 8 * hi;
                float y = acc[rt][t][r] + bv;
                Y[(size_t)grow * DM + gcol] = (_Float16)(y > 0.f ? y : 0.f);
            }
        }
    }
}

// ---------------------------------------------------------------------------
// Fused neighbor gather-sum + per-head softmax + head-mean. One block per
// (b,i). d = l*8 + h (head fastest). E matrices are f16, gathered with b128.
// ---------------------------------------------------------------------------
__global__ __launch_bounds__(128) void colight_attn_fused(
    const float* __restrict__ adj,        // [B,N,K,N] one-hot f32
    const _Float16* __restrict__ El,
    const _Float16* __restrict__ Ea,
    const _Float16* __restrict__ Eh,
    float* __restrict__ out)              // [B,N,128] f32
{
    const int bi  = blockIdx.x;           // b*128 + i
    const int b   = bi >> 7;
    const int tid = threadIdx.x;

    __shared__ int   idxs[8];
    __shared__ float Sh[1024];
    __shared__ float P[8][128];           // logits -> probs [head][l]
    __shared__ float red[128];

    // 1. recover neighbor indices from one-hot rows (entries exactly 0/1)
    #pragma unroll
    for (int k = 0; k < 8; ++k) {
        float v = adj[(((size_t)bi * 8) + k) * 128 + tid];
        if (v > 0.5f) idxs[k] = tid;
    }
    __syncthreads();

    int rows[8];
    #pragma unroll
    for (int k = 0; k < 8; ++k) rows[k] = b * 128 + idxs[k];

    // 2+3. gather-sum 8 rows of Ea/Eh (b128 loads); logits in one pass.
    const int d0 = tid * 8;
    float sa[8] = {}, sh[8] = {};
    #pragma unroll
    for (int k = 0; k < 8; ++k) {
        const v8h ea = *(const v8h*)(Ea + (size_t)rows[k] * DM + d0);
        const v8h eh = *(const v8h*)(Eh + (size_t)rows[k] * DM + d0);
        #pragma unroll
        for (int j = 0; j < 8; ++j) {
            sa[j] += (float)ea[j];
            sh[j] += (float)eh[j];
        }
    }
    const v8h ag = *(const v8h*)(El + (size_t)bi * DM + d0);
    #pragma unroll
    for (int j = 0; j < 8; ++j) {          // d0%8==0: head=j, l=tid
        Sh[d0 + j] = sh[j];
        P[j][tid]  = (float)ag[j] * sa[j];
    }
    __syncthreads();

    // 4. softmax over l per head; thread = h*16 + j16 owns 8 l's
    const int h = tid >> 4;
    const int j16 = tid & 15;
    float lmax = -3.0e38f;
    #pragma unroll
    for (int q = 0; q < 8; ++q) lmax = fmaxf(lmax, P[h][j16 * 8 + q]);
    red[tid] = lmax;
    __syncthreads();
    float hmax = -3.0e38f;
    #pragma unroll
    for (int q = 0; q < 16; ++q) hmax = fmaxf(hmax, red[h * 16 + q]);
    __syncthreads();

    float lsum = 0.f;
    #pragma unroll
    for (int q = 0; q < 8; ++q) {
        float e = __expf(P[h][j16 * 8 + q] - hmax);
        P[h][j16 * 8 + q] = e;
        lsum += e;
    }
    red[tid] = lsum;
    __syncthreads();
    float hsum = 0.f;
    #pragma unroll
    for (int q = 0; q < 16; ++q) hsum += red[h * 16 + q];
    const float inv = 1.0f / hsum;
    #pragma unroll
    for (int q = 0; q < 8; ++q) P[h][j16 * 8 + q] *= inv;
    __syncthreads();

    // 5. out[bi][l] = (1/8) * sum_h P[h][l] * Sh[l*8 + h]
    const int l = tid;
    float accv = 0.f;
    #pragma unroll
    for (int hh = 0; hh < 8; ++hh)
        accv += P[hh][l] * Sh[l * 8 + hh];
    out[(size_t)bi * 128 + l] = accv * 0.125f;
}

// ---------------------------------------------------------------------------
// Workspace (f16 units): Xh | T0 | T1 | T2 | El | Ea | Eh = 7 x 2MB = 14MB.
// ---------------------------------------------------------------------------
extern "C" void kernel_launch(void* const* d_in, const int* in_sizes, int n_in,
                              void* d_out, int out_size, void* d_ws, size_t ws_size,
                              hipStream_t stream) {
    const float* embedded = (const float*)d_in[0];
    const float* adj      = (const float*)d_in[1];
    const float* Wl       = (const float*)d_in[2];
    const float* bl       = (const float*)d_in[3];
    const float* Wa       = (const float*)d_in[4];
    const float* ba       = (const float*)d_in[5];
    const float* Wh       = (const float*)d_in[6];
    const float* bh       = (const float*)d_in[7];
    float* out = (float*)d_out;

    _Float16* ws = (_Float16*)d_ws;
    _Float16* Xh = ws;
    _Float16* T0 = ws + (size_t)1 * MEGA;
    _Float16* T1 = ws + (size_t)2 * MEGA;
    _Float16* T2 = ws + (size_t)3 * MEGA;
    _Float16* El = ws + (size_t)4 * MEGA;
    _Float16* Ea = ws + (size_t)5 * MEGA;
    _Float16* Eh = ws + (size_t)6 * MEGA;

    cvt_f32_to_f16<<<512, 256, 0, stream>>>(embedded, Xh);
    transpose_to_f16<<<dim3(32, 32, 3), 256, 0, stream>>>(Wl, Wa, Wh, T0, T1, T2);

    gemm_bias_relu_wmma<<<dim3(DM / BN, DM / BM, 3), 128, 0, stream>>>(
        Xh, T0, bl, T1, ba, T2, bh, El, Ea, Eh);

    colight_attn_fused<<<1024, 128, 0, stream>>>(adj, El, Ea, Eh, out);
}
